// PINOBlock_84963043050103
// MI455X (gfx1250) — matrix operand order repaired and checked by reading, not compiled, not run on hardware
//
#include <hip/hip_runtime.h>
#include <hip/hip_bf16.h>
#include <math.h>

// ---------------- problem constants ----------------
#define BATCH 2
#define NPTS  (64*64*64)   // 262144 points per batch
#define FM    64           // F modes
#define CIN   32
#define COUT  32

typedef __attribute__((ext_vector_type(4)))  int    v4i;
typedef __attribute__((ext_vector_type(8)))  float  v8f;
typedef __attribute__((ext_vector_type(16))) __bf16 v16bf;

// Assemble a 16-bit A-matrix (16x32) fragment from LDS rows of 64 bytes:
// per lane, two ds_load_b128 at +0 and +16 bytes (16 contiguous K).
__device__ inline v16bf ld_frag_gap(const __bf16* lds, int byteOff) {
  const v4i* q = (const v4i*)((const char*)lds + byteOff);
  union { struct { v4i a, b; } s; v16bf v; } u;
  u.s.a = q[0];      // bytes [0,16)
  u.s.b = q[1];      // bytes [16,32)
  return u.v;
}
// Assemble a 16-bit B-matrix (32x16) fragment: 16 contiguous halves per lane.
__device__ inline v16bf ld_frag_seq(const __bf16* lds, int byteOff) {
  const v4i* q = (const v4i*)((const char*)lds + byteOff);
  union { struct { v4i a, b; } s; v16bf v; } u;
  u.s.a = q[0];
  u.s.b = q[1];
  return u.v;
}

__device__ inline v8f zero8() { v8f z = {0.f,0.f,0.f,0.f,0.f,0.f,0.f,0.f}; return z; }

__device__ inline v8f wmma_bf16(v16bf A, v16bf B, v8f C) {
  return __builtin_amdgcn_wmma_f32_16x16x32_bf16(false, A, false, B, (short)0, C,
                                                 false, false);
}

// Async global->LDS copy of 16 bytes per lane (ASYNCcnt-tracked, no VGPR
// round trip).  LDS byte address = low 32 bits of the flat pointer (the
// aperture logic truncates addr[31:0] to the LDS offset).
__device__ inline void async_copy_b128(void* lds_dst, const void* gsrc) {
  unsigned lo = (unsigned)(size_t)lds_dst;
  unsigned long long ga = (unsigned long long)(size_t)gsrc;
  asm volatile("global_load_async_to_lds_b128 %0, %1, off"
               :: "v"(lo), "v"(ga) : "memory");
}

// ---------------- kernel 0: zero AX accumulator ----------------
__global__ __launch_bounds__(256) void k0_zero(float* p, int n) {
  int i = blockIdx.x * 256 + threadIdx.x;
  if (i < n) p[i] = 0.0f;
}

// ---------------- kernel 1: AX[b,f,g] = sum_n A[n,f] * X[n,g] ----------------
// A = a @ modes_a^T (real), X = exp(-2*pi*i * x @ modes_x^T) (complex).
#define K1_PTS 2048
#define K1_IT  (K1_PTS/32)

__global__ __launch_bounds__(256) void k1_accum(
    const float* __restrict__ a, const float* __restrict__ x,
    const float* __restrict__ modes_a, const float* __restrict__ modes_x,
    float* __restrict__ AXre, float* __restrict__ AXim)
{
  __shared__ __align__(16) float  s_ma[FM*CIN];     // modes_a [f][ci]   8KB
  __shared__            float     s_mx[FM*3];       // modes_x [g][d]
  __shared__ __align__(16) float  s_x[K1_PTS*3];    // whole-block coords 24KB
  __shared__ __align__(16) float  s_a[2][32*CIN];   // double-buffered a  8KB
  __shared__ __align__(16) __bf16 sAT[FM*32];       // A^T [f][n]         4KB
  __shared__ __align__(16) __bf16 sXre[FM*32];      // [g][n] cos         4KB
  __shared__ __align__(16) __bf16 sXim[FM*32];      // [g][n] -sin        4KB

  const int t  = threadIdx.x;
  const int b  = blockIdx.y;
  const int n0 = blockIdx.x * K1_PTS;

  for (int i = t; i < FM*CIN; i += 256) s_ma[i] = modes_a[i];
  if (t < FM*3) s_mx[t] = modes_x[t];

  // -- async preload: all x for this block (6x b128/thread) + first a tile --
  {
    const char* gx = (const char*)(x + ((size_t)b*NPTS + n0) * 3);
    #pragma unroll
    for (int r = 0; r < 6; ++r)   // 1536 float4 == K1_PTS*3 floats
      async_copy_b128(&((float4*)s_x)[t + r*256], gx + (size_t)(t + r*256)*16);
    const char* ga0 = (const char*)(a + ((size_t)b*NPTS + n0) * CIN);
    async_copy_b128(&((float4*)s_a[0])[t], ga0 + (size_t)t*16);
    asm volatile("s_wait_asynccnt 0x0" ::: "memory");
  }
  __syncthreads();

  const int lane = t & 31;
  const int w    = t >> 5;
  const int fi   = w & 3;          // this wave's f-tile (same for both g-tiles)
  const int gi0  = w >> 2;         // g-tiles {gi0, gi0+2}
  const int lh   = (lane >= 16);   // half-wave select

  v8f accRe0 = zero8(), accIm0 = zero8(), accRe1 = zero8(), accIm1 = zero8();

  const int aoff  = (fi*16 + (lane & 15)) * 64 + lh * 32;          // into sAT
  const int boff0 = (gi0*16 + (lane & 15)) * 64 + lh * 32;         // into sX*
  const int boff1 = ((gi0+2)*16 + (lane & 15)) * 64 + lh * 32;

  for (int it = 0; it < K1_IT; ++it) {
    const int buf = it & 1;
    // -- kick off next a tile while we compute on the current one --
    if (it + 1 < K1_IT) {
      const char* gan =
          (const char*)(a + ((size_t)b*NPTS + n0 + (it+1)*32) * CIN);
      async_copy_b128(&((float4*)s_a[buf ^ 1])[t], gan + (size_t)t*16);
      asm volatile("s_wait_asynccnt 0x1" ::: "memory");  // current tile landed
    } else {
      asm volatile("s_wait_asynccnt 0x0" ::: "memory");
    }

    // -- compute A and X tiles into WMMA-friendly LDS layouts --
    {
      const int nl = t >> 3, grp = t & 7;
      float av[CIN];
      #pragma unroll
      for (int i = 0; i < CIN; ++i) av[i] = s_a[buf][nl*CIN + i];
      #pragma unroll
      for (int j = 0; j < 8; ++j) {
        const int f = grp*8 + j;
        float s = 0.f;
        #pragma unroll
        for (int i = 0; i < CIN; ++i) s += av[i] * s_ma[f*CIN + i];
        sAT[f*32 + nl] = (__bf16)s;
      }
      const int nx = (it*32 + nl) * 3;
      const float x0 = s_x[nx+0], x1 = s_x[nx+1], x2 = s_x[nx+2];
      #pragma unroll
      for (int j = 0; j < 8; ++j) {
        const int g = grp*8 + j;
        const float ph = x0*s_mx[g*3+0] + x1*s_mx[g*3+1] + x2*s_mx[g*3+2];
        float sn, cs;
        __sincosf(-6.283185307179586f * ph, &sn, &cs);   // exp(-2*pi*i*ph)
        sXre[g*32 + nl] = (__bf16)cs;
        sXim[g*32 + nl] = (__bf16)sn;                    // = -sin(2*pi*ph)
      }
    }
    __syncthreads();

    // -- WMMA: accumulate this wave's two (f,g) tiles over K=32 points --
    {
      v16bf A   = ld_frag_gap(sAT,  aoff);
      v16bf Br0 = ld_frag_seq(sXre, boff0);
      v16bf Bi0 = ld_frag_seq(sXim, boff0);
      v16bf Br1 = ld_frag_seq(sXre, boff1);
      v16bf Bi1 = ld_frag_seq(sXim, boff1);
      accRe0 = wmma_bf16(A, Br0, accRe0);
      accIm0 = wmma_bf16(A, Bi0, accIm0);
      accRe1 = wmma_bf16(A, Br1, accRe1);
      accIm1 = wmma_bf16(A, Bi1, accIm1);
    }
    __syncthreads();
  }

  // -- flush: f32 atomics into the tiny [B,F,F] accumulator --
  const int g0 = gi0*16 + (lane & 15);
  const int g1 = (gi0+2)*16 + (lane & 15);
  #pragma unroll
  for (int r = 0; r < 8; ++r) {
    const int f = fi*16 + r + 8*lh;
    atomicAdd(&AXre[((size_t)b*FM + f)*FM + g0], accRe0[r]);
    atomicAdd(&AXim[((size_t)b*FM + f)*FM + g0], accIm0[r]);
    atomicAdd(&AXre[((size_t)b*FM + f)*FM + g1], accRe1[r]);
    atomicAdd(&AXim[((size_t)b*FM + f)*FM + g1], accIm1[r]);
  }
}

// ---------------- kernel 2: apply AX, project, residual, GELU ----------------
// U[n,f] = (1/N) * sum_g (AXre[f,g]*Yre[n,g] - AXim[f,g]*Yim[n,g]),
// Y = exp(+2*pi*i * x @ modes_y^T)  (source passes x for both coord args);
// out = gelu(U @ modes_u + a @ fc_w + fc_b)
__global__ __launch_bounds__(256) void k2_apply(
    const float* __restrict__ a, const float* __restrict__ x,
    const float* __restrict__ modes_y, const float* __restrict__ modes_u,
    const float* __restrict__ fc_w, const float* __restrict__ fc_b,
    const float* __restrict__ AXre, const float* __restrict__ AXim,
    float* __restrict__ out)
{
  __shared__            float     s_my[FM*3];
  __shared__ __align__(16) float  s_x2[128*3];
  __shared__            float     s_fb[COUT];
  __shared__ __align__(16) __bf16 sYre[128*FM];    // [n][g] cos      16KB
  __shared__ __align__(16) __bf16 sYim[128*FM];    // [n][g] -sin     16KB
  __shared__ __align__(16) __bf16 sU  [128*FM];    // [n][f]          16KB
  __shared__ __align__(16) __bf16 sAbf[128*CIN];   // [n][ci]          8KB
  __shared__ __align__(16) __bf16 sXr [FM*FM];     // AXre/N [f][g]    8KB
  __shared__ __align__(16) __bf16 sXi [FM*FM];     //                  8KB
  __shared__ __align__(16) __bf16 sMu [COUT*FM];   // modes_u^T [o][f] 4KB
  __shared__ __align__(16) __bf16 sFw [COUT*CIN];  // fc_w^T   [o][ci] 2KB

  const int t  = threadIdx.x;
  const int b  = blockIdx.y;
  const int n0 = blockIdx.x * 128;
  const float invN = 1.0f / (float)NPTS;

  // -- cooperative once-per-block staging --
  if (t < FM*3)  s_my[t] = modes_y[t];
  if (t < COUT)  s_fb[t] = fc_b[t];
  for (int i = t; i < 128*3; i += 256) s_x2[i] = x[((size_t)b*NPTS + n0)*3 + i];
  {
    const float4* ga = (const float4*)(a + ((size_t)b*NPTS + n0) * CIN);
    #pragma unroll
    for (int r = 0; r < 4; ++r) {
      const float4 v = ga[t + r*256];
      const int idx = (t + r*256) * 4;
      sAbf[idx+0] = (__bf16)v.x;  sAbf[idx+1] = (__bf16)v.y;
      sAbf[idx+2] = (__bf16)v.z;  sAbf[idx+3] = (__bf16)v.w;
    }
  }
  for (int i = t; i < FM*FM; i += 256) {
    sXr[i] = (__bf16)(AXre[(size_t)b*FM*FM + i] * invN);  // fold 1/N here
    sXi[i] = (__bf16)(AXim[(size_t)b*FM*FM + i] * invN);
  }
  for (int i = t; i < FM*COUT; i += 256) {   // transpose modes_u -> [o][f]
    const int f = i >> 5, o = i & 31;
    sMu[o*FM + f] = (__bf16)modes_u[i];
  }
  for (int i = t; i < CIN*COUT; i += 256) {  // transpose fc_w -> [o][ci]
    const int ci = i >> 5, o = i & 31;
    sFw[o*CIN + ci] = (__bf16)fc_w[i];
  }
  __syncthreads();

  // -- Fourier features Y (store -sin so no NEG modifiers needed) --
  {
    const int nl = t & 127, gh = (t >> 7) * 32;
    const float x0 = s_x2[nl*3+0], x1 = s_x2[nl*3+1], x2 = s_x2[nl*3+2];
    for (int j = 0; j < 32; ++j) {
      const int g = gh + j;
      const float ph = x0*s_my[g*3+0] + x1*s_my[g*3+1] + x2*s_my[g*3+2];
      float sn, cs;
      __sincosf(6.283185307179586f * ph, &sn, &cs);      // exp(+2*pi*i*ph)
      sYre[nl*FM + g] = (__bf16)cs;
      sYim[nl*FM + g] = (__bf16)(-sn);
    }
  }
  __syncthreads();

  const int lane = t & 31;
  const int w    = t >> 5;        // each wave owns 16 points
  const int lh   = (lane >= 16);
  const int nrow = w*16 + (lane & 15);

  // -- U = Yre*AXre + (-Yim)*AXim : 4 f-tiles, K over g in two 32-chunks --
  v8f Uacc[4];
  #pragma unroll
  for (int fi = 0; fi < 4; ++fi) Uacc[fi] = zero8();
  #pragma unroll
  for (int c = 0; c < 2; ++c) {
    const int abase = nrow*128 + c*64 + lh*32;           // 128B rows
    v16bf Yr = ld_frag_gap(sYre, abase);
    v16bf Yi = ld_frag_gap(sYim, abase);
    #pragma unroll
    for (int fi = 0; fi < 4; ++fi) {
      const int bbase = (fi*16 + (lane & 15))*128 + c*64 + lh*32;
      v16bf Br = ld_frag_seq(sXr, bbase);
      v16bf Bi = ld_frag_seq(sXi, bbase);
      Uacc[fi] = wmma_bf16(Yr, Br, Uacc[fi]);
      Uacc[fi] = wmma_bf16(Yi, Bi, Uacc[fi]);
    }
  }
  // spill U to LDS (bf16) for the next GEMM's A-fragment layout
  #pragma unroll
  for (int fi = 0; fi < 4; ++fi) {
    const int f = fi*16 + (lane & 15);
    #pragma unroll
    for (int r = 0; r < 8; ++r) {
      const int n = w*16 + r + 8*lh;
      sU[n*FM + f] = (__bf16)Uacc[fi][r];
    }
  }
  __syncthreads();

  // -- u = U @ modes_u + a @ fc_w : 2 o-tiles --
  v8f uacc[2];
  uacc[0] = zero8(); uacc[1] = zero8();
  #pragma unroll
  for (int c = 0; c < 2; ++c) {
    v16bf Ua = ld_frag_gap(sU, nrow*128 + c*64 + lh*32);
    #pragma unroll
    for (int oi = 0; oi < 2; ++oi) {
      v16bf Bo = ld_frag_seq(sMu, (oi*16 + (lane & 15))*128 + c*64 + lh*32);
      uacc[oi] = wmma_bf16(Ua, Bo, uacc[oi]);
    }
  }
  {
    v16bf Aa = ld_frag_gap(sAbf, nrow*64 + lh*32);       // 64B rows
    #pragma unroll
    for (int oi = 0; oi < 2; ++oi) {
      v16bf Bw = ld_frag_seq(sFw, (oi*16 + (lane & 15))*64 + lh*32);
      uacc[oi] = wmma_bf16(Aa, Bw, uacc[oi]);
    }
  }

  // -- + bias, tanh-GELU, store --
  #pragma unroll
  for (int oi = 0; oi < 2; ++oi) {
    const int o = oi*16 + (lane & 15);
    const float bsv = s_fb[o];
    #pragma unroll
    for (int r = 0; r < 8; ++r) {
      const int n = n0 + w*16 + r + 8*lh;
      const float h = uacc[oi][r] + bsv;
      const float g3 = 0.7978845608028654f * (h + 0.044715f * h * h * h);
      out[((size_t)b*NPTS + n)*COUT + o] = 0.5f * h * (1.0f + tanhf(g3));
    }
  }
}

// ---------------- launcher ----------------
extern "C" void kernel_launch(void* const* d_in, const int* in_sizes, int n_in,
                              void* d_out, int out_size, void* d_ws, size_t ws_size,
                              hipStream_t stream) {
  const float* a       = (const float*)d_in[0];
  const float* x       = (const float*)d_in[1];
  // d_in[2] = y : unused (reference calls att(h, x, x))
  const float* modes_a = (const float*)d_in[3];
  const float* modes_x = (const float*)d_in[4];
  const float* modes_y = (const float*)d_in[5];
  const float* modes_u = (const float*)d_in[6];
  const float* fc_w    = (const float*)d_in[7];
  const float* fc_b    = (const float*)d_in[8];
  float* out  = (float*)d_out;

  float* AXre = (float*)d_ws;                 // [B][F][F]
  float* AXim = AXre + (size_t)BATCH*FM*FM;   // [B][F][F]
  const int axTotal = 2 * BATCH * FM * FM;    // 16384 floats

  k0_zero<<<dim3((axTotal + 255)/256), dim3(256), 0, stream>>>(AXre, axTotal);

  k1_accum<<<dim3(NPTS / K1_PTS, BATCH), dim3(256), 0, stream>>>(
      a, x, modes_a, modes_x, AXre, AXim);

  k2_apply<<<dim3(NPTS / 128, BATCH), dim3(256), 0, stream>>>(
      a, x, modes_y, modes_u, fc_w, fc_b, AXre, AXim, out);
}
